// STFSModule_76124000354390
// MI455X (gfx1250) — matrix operand order repaired and checked by reading, not compiled, not run on hardware
//
#include <hip/hip_runtime.h>

// ---------------- types ----------------
typedef __bf16 bf16_t;
typedef bf16_t v16bf __attribute__((ext_vector_type(16)));
typedef float  v8f   __attribute__((ext_vector_type(8)));
typedef unsigned int u32x4 __attribute__((ext_vector_type(4)));
typedef float  f32x4 __attribute__((ext_vector_type(4)));
typedef bf16_t bf16x4 __attribute__((ext_vector_type(4)));

// ---------------- problem constants ----------------
constexpr int KK   = 1024;          // RoI pairs
constexpr int CC   = 128;           // channels
constexpr int RR   = 7;
constexpr int DD   = CC * RR * RR;  // 6272
constexpr int FFN  = 1024;
constexpr int NW   = 36864;         // C*(C/GROUPS)*9 dyn-weight columns
constexpr int MM   = 64;            // boxes
constexpr size_t KD = (size_t)KK * DD;

// ---------------- workspace layout (bytes, all 256-aligned) ----------------
constexpr size_t OFF_BIGA  = 0;                       // wv_b -> wo_b -> dynw (78,675,968)
constexpr size_t OFF_RFB   = 78675968;                // rf bf16 -> later dcb   (12,845,056)
constexpr size_t OFF_VB    = 91521024;                // v bf16  -> later wf1b  (12,845,056)
constexpr size_t OFF_ATTN  = 104366080;               // attn f32 -> later lin  (25,690,112)
constexpr size_t OFF_X     = 130056192;               // x f32   -> later wf2b  (25,690,112)
constexpr size_t OFF_GAP   = 155746304;               // gap f32 (524,288)
constexpr size_t OFF_GAPB  = 156270592;               // gap bf16 (262,144)
constexpr size_t OFF_WGENB = 156532736;               // wgen bf16 -> later hb  (9,437,184)
constexpr size_t OFF_DC    = 165969920;               // dc f32 (25,690,112)

// ---------------- helpers ----------------
__device__ inline float blockReduceSum(float v, float* red) {
    for (int m = 16; m; m >>= 1) v += __shfl_xor(v, m, 32);
    int wave = threadIdx.x >> 5, lane = threadIdx.x & 31;
    if (lane == 0) red[wave] = v;
    __syncthreads();
    float r = (threadIdx.x < 8) ? red[threadIdx.x] : 0.f;
    if (wave == 0) {
        for (int m = 16; m; m >>= 1) r += __shfl_xor(r, m, 32);
        if (lane == 0) red[0] = r;
    }
    __syncthreads();
    float out = red[0];
    __syncthreads();
    return out;
}

// ---------------- f32 -> bf16 (vectorized x4) ----------------
__global__ __launch_bounds__(256) void cvt_f32_bf16(const float* __restrict__ in,
                                                    bf16_t* __restrict__ out, int n4) {
    int i = blockIdx.x * 256 + threadIdx.x;
    if (i >= n4) return;
    f32x4 v = ((const f32x4*)in)[i];
    bf16x4 o;
    o[0] = (bf16_t)v[0]; o[1] = (bf16_t)v[1]; o[2] = (bf16_t)v[2]; o[3] = (bf16_t)v[3];
    ((bf16x4*)out)[i] = o;
}

// ---------------- bf16 WMMA GEMM: C[M,N] = A[M,K] * W[N,K]^T + bias ----------------
// Block = 8 waves arranged 4(M) x 2(N); block tile 128x128.
// Each wave computes 32(M) x 64(N): 2x4 accumulators, 8 WMMAs per 12 b128 loads.
enum { EPI_F32_BIAS = 0, EPI_BF16_BIAS = 1, EPI_BF16_BIAS_RELU = 2 };

template <int EPI>
__global__ __launch_bounds__(256, 2) void gemm_bf16(
    const bf16_t* __restrict__ A, const bf16_t* __restrict__ B,
    const float* __restrict__ bias, float* __restrict__ outF,
    bf16_t* __restrict__ outB, int M, int N, int Kd) {
    union Frag { v16bf v; u32x4 q[2]; };
    const int lane = threadIdx.x & 31;
    const int wave = threadIdx.x >> 5;
    const int m0 = blockIdx.y * 128 + (wave >> 1) * 32;
    const int n0 = blockIdx.x * 128 + (wave & 1) * 64;
    const int lh = lane >> 4;     // half-wave select
    const int ll = lane & 15;

    v8f acc[2][4];
#pragma unroll
    for (int i = 0; i < 2; ++i)
#pragma unroll
        for (int j = 0; j < 4; ++j)
#pragma unroll
            for (int e = 0; e < 8; ++e) acc[i][j][e] = 0.f;

    const bf16_t* aR[2];
    const bf16_t* bR[4];
#pragma unroll
    for (int i = 0; i < 2; ++i) aR[i] = A + (size_t)(m0 + 16 * i + ll) * Kd;
#pragma unroll
    for (int j = 0; j < 4; ++j) bR[j] = B + (size_t)(n0 + 16 * j + ll) * Kd;
    const int aOff = lh * 8;      // A: K = {e<8: e, e>=8: e+8} + 8*lh
    const int bOff = lh * 8;      // B: K = {e<8: e, e>=8: e+8} + 8*lh

    for (int kb = 0; kb < Kd; kb += 32) {
        Frag a[2], b[4];
#pragma unroll
        for (int i = 0; i < 2; ++i) {
            a[i].q[0] = *(const u32x4*)(aR[i] + kb + aOff);
            a[i].q[1] = *(const u32x4*)(aR[i] + kb + 16 + aOff);
        }
#pragma unroll
        for (int j = 0; j < 4; ++j) {
            b[j].q[0] = *(const u32x4*)(bR[j] + kb + bOff);
            b[j].q[1] = *(const u32x4*)(bR[j] + kb + 16 + bOff);
        }
#pragma unroll
        for (int i = 0; i < 2; ++i)
#pragma unroll
            for (int j = 0; j < 4; ++j)
                acc[i][j] = __builtin_amdgcn_wmma_f32_16x16x32_bf16(
                    false, a[i].v, false, b[j].v, (short)0, acc[i][j], false, false);
    }

#pragma unroll
    for (int j = 0; j < 4; ++j) {
        const int n = n0 + j * 16 + ll;
        const float bv = bias[n];
#pragma unroll
        for (int i = 0; i < 2; ++i) {
#pragma unroll
            for (int e = 0; e < 8; ++e) {
                const int m = m0 + i * 16 + e + lh * 8;
                float val = acc[i][j][e] + bv;
                if (EPI == EPI_F32_BIAS) {
                    outF[(size_t)m * N + n] = val;
                } else if (EPI == EPI_BF16_BIAS) {
                    outB[(size_t)m * N + n] = (bf16_t)val;
                } else {
                    outB[(size_t)m * N + n] = (bf16_t)fmaxf(val, 0.f);
                }
            }
        }
    }
}

// ---------------- LN1 (+residual) and GAP ----------------
__global__ __launch_bounds__(256) void ln1_gap(const float* __restrict__ attn,
                                               const float* __restrict__ wf,
                                               const float* __restrict__ g1,
                                               const float* __restrict__ b1,
                                               float* __restrict__ x,
                                               float* __restrict__ gap,
                                               bf16_t* __restrict__ gapb) {
    __shared__ float buf[DD];
    __shared__ float red[16];
    const int r = blockIdx.x, t = threadIdx.x;
    const float* pa = attn + (size_t)r * DD;
    const float* pw = wf + (size_t)r * DD;
    float s = 0.f, s2 = 0.f;
    for (int d = t; d < DD; d += 256) {
        float v = pa[d] + pw[d];
        buf[d] = v; s += v; s2 += v * v;
    }
    float S = blockReduceSum(s, red);
    float S2 = blockReduceSum(s2, red);
    const float mu = S * (1.f / DD);
    const float rstd = rsqrtf(S2 * (1.f / DD) - mu * mu + 1e-5f);
    float* px = x + (size_t)r * DD;
    for (int d = t; d < DD; d += 256) {
        float xn = (buf[d] - mu) * rstd * g1[d] + b1[d];
        px[d] = xn; buf[d] = xn;
    }
    __syncthreads();
    if (t < CC) {
        const float* pc = buf + t * 49;
        float sum = 0.f;
#pragma unroll
        for (int i = 0; i < 49; ++i) sum += pc[i];
        float g = sum * (1.f / 49.f);
        gap[r * CC + t] = g;
        gapb[r * CC + t] = (bf16_t)g;
    }
}

// ---------------- per-sample grouped dynamic conv + residual + LN2 ----------------
__global__ __launch_bounds__(256) void dynconv_ln2(const float* __restrict__ x,
                                                   const bf16_t* __restrict__ dynw,
                                                   const float* __restrict__ g2,
                                                   const float* __restrict__ b2,
                                                   float* __restrict__ dc,
                                                   bf16_t* __restrict__ dcb) {
    extern __shared__ char smem[];
    float* xin = (float*)smem;                        // 6272 f32
    bf16_t* wds = (bf16_t*)(smem + DD * 4);           // 36864 bf16
    __shared__ float red[16];
    const int k = blockIdx.x, t = threadIdx.x;
    const float* px = x + (size_t)k * DD;
    for (int d = t; d < DD; d += 256) xin[d] = px[d];
    const unsigned int* wsrc = (const unsigned int*)(dynw + (size_t)k * NW);
    unsigned int* wdst = (unsigned int*)wds;
    for (int i = t; i < NW / 2; i += 256) wdst[i] = wsrc[i];
    __syncthreads();

    float acc[25];
#pragma unroll
    for (int i = 0; i < 25; ++i) acc[i] = 0.f;
#pragma unroll 1
    for (int i = 0; i < 25; ++i) {
        const int o = t + i * 256;
        if (o >= DD) break;
        const int co = o / 49;
        const int pos = o - co * 49;
        const int y = pos / 7, xx = pos - (pos / 7) * 7;
        const bf16_t* wp = wds + co * 288;            // 32 ci * 9 taps
        const float* ip = xin + ((co >> 5) << 5) * 49;
        float s = 0.f;
#pragma unroll
        for (int dy = 0; dy < 3; ++dy) {
            const int iy = y + dy - 1;
            if ((unsigned)iy >= 7u) continue;
#pragma unroll
            for (int dx = 0; dx < 3; ++dx) {
                const int ix = xx + dx - 1;
                if ((unsigned)ix >= 7u) continue;
                const bf16_t* wpp = wp + dy * 3 + dx;
                const float* ipp = ip + iy * 7 + ix;
#pragma unroll 8
                for (int ci = 0; ci < 32; ++ci)
                    s += (float)wpp[ci * 9] * ipp[ci * 49];
            }
        }
        acc[i] = s;
    }

    float ssum = 0.f, ssq = 0.f;
#pragma unroll 1
    for (int i = 0; i < 25; ++i) {
        const int o = t + i * 256;
        if (o >= DD) break;
        const float rv = acc[i] + xin[o];
        ssum += rv; ssq += rv * rv;
    }
    const float S = blockReduceSum(ssum, red);
    const float S2 = blockReduceSum(ssq, red);
    const float mu = S * (1.f / DD);
    const float rstd = rsqrtf(S2 * (1.f / DD) - mu * mu + 1e-5f);
    float* pdc = dc + (size_t)k * DD;
    bf16_t* pdb = dcb + (size_t)k * DD;
#pragma unroll 1
    for (int i = 0; i < 25; ++i) {
        const int o = t + i * 256;
        if (o >= DD) break;
        const float rv = acc[i] + xin[o];
        const float val = (rv - mu) * rstd * g2[o] + b2[o];
        pdc[o] = val; pdb[o] = (bf16_t)val;
    }
}

// ---------------- LN3 (+residual) -> final output ----------------
__global__ __launch_bounds__(256) void ln3(const float* __restrict__ lin,
                                           const float* __restrict__ dc,
                                           const float* __restrict__ g3,
                                           const float* __restrict__ b3,
                                           float* __restrict__ out) {
    __shared__ float buf[DD];
    __shared__ float red[16];
    const int r = blockIdx.x, t = threadIdx.x;
    const float* pl = lin + (size_t)r * DD;
    const float* pd = dc + (size_t)r * DD;
    float s = 0.f, s2 = 0.f;
    for (int d = t; d < DD; d += 256) {
        float v = pl[d] + pd[d];
        buf[d] = v; s += v; s2 += v * v;
    }
    float S = blockReduceSum(s, red);
    float S2 = blockReduceSum(s2, red);
    const float mu = S * (1.f / DD);
    const float rstd = rsqrtf(S2 * (1.f / DD) - mu * mu + 1e-5f);
    float* po = out + (size_t)r * DD;
    for (int d = t; d < DD; d += 256)
        po[d] = (buf[d] - mu) * rstd * g3[d] + b3[d];
}

// ---------------- box IoU / L1-distance cost matrix (64x64) ----------------
__global__ __launch_bounds__(256) void cost_kernel(const float* __restrict__ refb,
                                                   const float* __restrict__ curb,
                                                   float* __restrict__ cost) {
    __shared__ float rB[MM * 4], cB[MM * 4];
    __shared__ float di[MM * MM], io[MM * MM];
    __shared__ float red[16];
    const int t = threadIdx.x;
    rB[t] = refb[t]; cB[t] = curb[t];
    __syncthreads();
    float lmax = 0.f;
    for (int p = t; p < MM * MM; p += 256) {
        const int i = p >> 6, j = p & 63;
        const float r0 = rB[i * 4], r1 = rB[i * 4 + 1], r2 = rB[i * 4 + 2], r3 = rB[i * 4 + 3];
        const float c0 = cB[j * 4], c1 = cB[j * 4 + 1], c2 = cB[j * 4 + 2], c3 = cB[j * 4 + 3];
        const float areaR = (r2 - r0) * (r3 - r1), areaC = (c2 - c0) * (c3 - c1);
        const float w = fmaxf(fminf(r2, c2) - fmaxf(r0, c0), 0.f);
        const float h = fmaxf(fminf(r3, c3) - fmaxf(r1, c1), 0.f);
        const float inter = w * h;
        const float iou = inter / (areaR + areaC - inter);
        const float dist = fabsf((r0 + r2) * 0.5f - (c0 + c2) * 0.5f) +
                           fabsf((r1 + r3) * 0.5f - (c1 + c3) * 0.5f);
        di[p] = dist; io[p] = iou;
        lmax = fmaxf(lmax, dist);
    }
    for (int m = 16; m; m >>= 1) lmax = fmaxf(lmax, __shfl_xor(lmax, m, 32));
    const int wave = t >> 5, lane = t & 31;
    if (lane == 0) red[wave] = lmax;
    __syncthreads();
    float gm = (t < 8) ? red[t] : 0.f;
    if (wave == 0) {
        for (int m = 16; m; m >>= 1) gm = fmaxf(gm, __shfl_xor(gm, m, 32));
        if (lane == 0) red[0] = gm;
    }
    __syncthreads();
    const float denom = fmaxf(red[0], 1.0f);
    for (int p = t; p < MM * MM; p += 256)
        cost[p] = -io[p] + 0.5f * di[p] / denom;
}

// ---------------- host orchestration ----------------
static inline void cvt(const float* in, bf16_t* out, size_t n, hipStream_t s) {
    int n4 = (int)(n / 4);
    cvt_f32_bf16<<<(n4 + 255) / 256, 256, 0, s>>>(in, out, n4);
}

extern "C" void kernel_launch(void* const* d_in, const int* in_sizes, int n_in,
                              void* d_out, int out_size, void* d_ws, size_t ws_size,
                              hipStream_t stream) {
    const float* wf     = (const float*)d_in[0];   // wrong_features (residual source)
    const float* rf     = (const float*)d_in[1];   // right_features
    const float* refbox = (const float*)d_in[2];
    const float* curbox = (const float*)d_in[3];
    // d_in[4..7] = wq,bq,wk,bk — algebraically eliminated (softmax over 1 key == 1)
    const float* wv = (const float*)d_in[8];  const float* bv = (const float*)d_in[9];
    const float* wo = (const float*)d_in[10]; const float* bo = (const float*)d_in[11];
    const float* g1 = (const float*)d_in[12]; const float* b1 = (const float*)d_in[13];
    const float* wgen_w = (const float*)d_in[14]; const float* wgen_b = (const float*)d_in[15];
    const float* g2 = (const float*)d_in[16]; const float* b2 = (const float*)d_in[17];
    const float* wf1 = (const float*)d_in[18]; const float* bf1 = (const float*)d_in[19];
    const float* wf2 = (const float*)d_in[20]; const float* bf2 = (const float*)d_in[21];
    const float* g3 = (const float*)d_in[22]; const float* b3 = (const float*)d_in[23];

    char* ws = (char*)d_ws;
    bf16_t* bigA  = (bf16_t*)(ws + OFF_BIGA);   // wv_b -> wo_b -> dynw
    bf16_t* rfb   = (bf16_t*)(ws + OFF_RFB);
    bf16_t* dcb   = (bf16_t*)(ws + OFF_RFB);    // reuse after rfb dead
    bf16_t* vb    = (bf16_t*)(ws + OFF_VB);
    bf16_t* wf1b  = (bf16_t*)(ws + OFF_VB);     // reuse after vb dead
    float*  attn  = (float*)(ws + OFF_ATTN);
    float*  lin   = (float*)(ws + OFF_ATTN);    // reuse after attn dead
    float*  x     = (float*)(ws + OFF_X);
    bf16_t* wf2b  = (bf16_t*)(ws + OFF_X);      // reuse after x dead
    float*  gap   = (float*)(ws + OFF_GAP);
    bf16_t* gapb  = (bf16_t*)(ws + OFF_GAPB);
    bf16_t* wgenb = (bf16_t*)(ws + OFF_WGENB);
    bf16_t* hb    = (bf16_t*)(ws + OFF_WGENB);  // reuse after wgenb dead
    float*  dc    = (float*)(ws + OFF_DC);

    float* out_agg  = (float*)d_out;
    float* out_cost = out_agg + KD;

    // 1) V projection: v = rf @ wv^T + bv  (bf16 out; attn == v since softmax(1)==1)
    cvt(rf, rfb, KD, stream);
    cvt(wv, bigA, (size_t)DD * DD, stream);
    {
        dim3 g(DD / 128, KK / 128);
        gemm_bf16<EPI_BF16_BIAS><<<g, 256, 0, stream>>>(rfb, bigA, bv, nullptr, vb, KK, DD, DD);
    }
    // 2) O projection: attn_out = v @ wo^T + bo (f32 out)
    cvt(wo, bigA, (size_t)DD * DD, stream);
    {
        dim3 g(DD / 128, KK / 128);
        gemm_bf16<EPI_F32_BIAS><<<g, 256, 0, stream>>>(vb, bigA, bo, attn, nullptr, KK, DD, DD);
    }
    // 3) LN1(attn + wf) -> x, and GAP(x) -> gap/gapb
    ln1_gap<<<KK, 256, 0, stream>>>(attn, wf, g1, b1, x, gap, gapb);
    // 4) dyn-weight gen: dynw = gap @ wgen_w^T + wgen_b (bf16 out)
    cvt(wgen_w, wgenb, (size_t)NW * CC, stream);
    {
        dim3 g(NW / 128, KK / 128);
        gemm_bf16<EPI_BF16_BIAS><<<g, 256, 0, stream>>>(gapb, wgenb, wgen_b, nullptr, bigA, KK, NW, CC);
    }
    // 5) per-sample grouped 3x3 conv + residual + LN2 -> dc / dcb
    {
        size_t lds = (size_t)DD * 4 + (size_t)NW * 2;  // 98,816 B
        dynconv_ln2<<<KK, 256, lds, stream>>>(x, bigA, g2, b2, dc, dcb);
    }
    // 6) FFN: h = relu(dc @ wf1^T + bf1); lin = h @ wf2^T + bf2
    cvt(wf1, wf1b, (size_t)FFN * DD, stream);
    cvt(wf2, wf2b, (size_t)DD * FFN, stream);
    {
        dim3 g(FFN / 128, KK / 128);
        gemm_bf16<EPI_BF16_BIAS_RELU><<<g, 256, 0, stream>>>(dcb, wf1b, bf1, nullptr, hb, KK, FFN, DD);
    }
    {
        dim3 g(DD / 128, KK / 128);
        gemm_bf16<EPI_F32_BIAS><<<g, 256, 0, stream>>>(hb, wf2b, bf2, lin, nullptr, KK, DD, FFN);
    }
    // 7) LN3(lin + dc) -> aggregated output
    ln3<<<KK, 256, 0, stream>>>(lin, dc, g3, b3, out_agg);
    // 8) Hungarian cost matrix
    cost_kernel<<<1, 256, 0, stream>>>(refbox, curbox, out_cost);
}
